// Decoder_90623809945905
// MI455X (gfx1250) — matrix operand order repaired and checked by reading, not compiled, not run on hardware
//
#include <hip/hip_runtime.h>
#include <hip/hip_bf16.h>

typedef _Float16 half_t;
typedef half_t v16h __attribute__((ext_vector_type(16)));
typedef half_t v8h  __attribute__((ext_vector_type(8)));
typedef float  v8f  __attribute__((ext_vector_type(8)));

// ---------------------------------------------------------------------------
// Weight pack: OIHW fp32 -> per-lane B-fragment image for v_wmma_f32_16x16x32_f16
// bp[nblk][t][lane][i]; K-order = tap*64 + c; B element i of lane l holds
// K = t*32 + ((l>=16)?8:0) + (i&7) + ((i>=8)?16:0), N = nblk*16 + (l&15).
// ---------------------------------------------------------------------------
__global__ void pack_weights(const float* __restrict__ w, half_t* __restrict__ bp, int Cout) {
    const int KT = 18; // (64*9)/32
    int tid = blockIdx.x * blockDim.x + threadIdx.x;
    int total = Cout * 576;
    if (tid >= total) return;
    int i    = tid & 15;
    int lane = (tid >> 4) & 31;
    int t    = (tid >> 9) % KT;
    int nblk = tid / (KT * 512);
    int Kg  = t * 32 + ((lane >= 16) ? 8 : 0) + (i & 7) + ((i >= 8) ? 16 : 0);
    int tap = Kg >> 6;
    int c   = Kg & 63;
    int N   = nblk * 16 + (lane & 15);
    bp[tid] = (half_t)w[((size_t)(N * 64 + c)) * 9 + tap];
}

// NCHW fp32 -> NHWC f16 with 1-pixel halo (hid staging only; small)
__global__ void nchw_to_nhwc_f16_halo(const float* __restrict__ src, half_t* __restrict__ dst,
                                      int B, int C, int H, int W) {
    size_t tid = (size_t)blockIdx.x * blockDim.x + threadIdx.x;
    size_t total = (size_t)B * C * H * W;
    if (tid >= total) return;
    int c = (int)(tid % C);
    size_t pix = tid / C;
    int x = (int)(pix % W);
    size_t t2 = pix / W;
    int y = (int)(t2 % H);
    int b = (int)(t2 / H);
    dst[(((size_t)b * (H + 2) + (y + 1)) * (W + 2) + (x + 1)) * C + c] =
        (half_t)src[(((size_t)b * C + c) * H + y) * W + x];
}

// ---------------------------------------------------------------------------
// Implicit-GEMM 3x3 conv (pad=1, stride=1), Cin=64, f16 WMMA, fp32 accumulate.
// Input is halo-padded NHWC f16 ((H+2)x(W+2), zero border), so ALL tap loads
// are unconditional: no predicates, no zero-fills in the hot loop.
// One wave = 16 pixels x 16 out-channels; 9 taps x 2 k-tiles = 18 WMMAs.
// Fuses bias, optional PixelShuffle(2) store, 64-way banked GN stats.
// ---------------------------------------------------------------------------
template<int COUT, int H, int W, bool SHUF>
__global__ __launch_bounds__(128)
void conv3x3_wmma(const half_t* __restrict__ act,    // [B,H+2,W+2,64] f16, zero halo
                  const half_t* __restrict__ bpack,  // [COUT/16][18][32][16] f16
                  const float*  __restrict__ bias,   // [COUT]
                  float*        __restrict__ raw,    // NCHW, post-shuffle dims
                  float*        __restrict__ stats)  // [B*2][64 banks][2]
{
    constexpr int CIN = 64;
    constexpr int HH = H + 2, WW = W + 2;
    const int lane = threadIdx.x & 31;
    const int wv   = threadIdx.x >> 5;
    const int nblk = blockIdx.y * 4 + wv;        // out-channel block of 16
    const int m0   = blockIdx.x * 16;            // 16 consecutive pixels, same image
    const int b    = m0 / (H * W);
    const int base = m0 % (H * W);

    // A-fragment addressing for this lane (row M = lane&15)
    const int idxA = base + (lane & 15);
    const int ya   = idxA / W;                   // interior coords
    const int xa   = idxA % W;
    const int kA   = (lane >= 16) ? 8 : 0;       // A-layout K sub-base

    const half_t* bp = bpack + (size_t)nblk * 18 * 512 + (size_t)lane * 16;

    v8f acc = {};
    #pragma unroll
    for (int ky = 0; ky < 3; ++ky) {
        // halo coords: interior (ya+ky-1) -> halo row (ya+ky); same for x
        const half_t* rp = act + (((size_t)b * HH + (ya + ky)) * WW + xa) * CIN + kA;
        #pragma unroll
        for (int kx = 0; kx < 3; ++kx) {
            const half_t* p = rp + kx * CIN;
            v8h q0 = *reinterpret_cast<const v8h*>(p);
            v8h q1 = *reinterpret_cast<const v8h*>(p + 16);
            v8h q2 = *reinterpret_cast<const v8h*>(p + 32);
            v8h q3 = *reinterpret_cast<const v8h*>(p + 48);
            const int t0 = (ky * 3 + kx) * 2;
            v16h a0 = __builtin_shufflevector(q0, q1, 0,1,2,3,4,5,6,7,8,9,10,11,12,13,14,15);
            v16h a1 = __builtin_shufflevector(q2, q3, 0,1,2,3,4,5,6,7,8,9,10,11,12,13,14,15);
            v16h b0 = *reinterpret_cast<const v16h*>(bp + (size_t)t0 * 512);
            v16h b1 = *reinterpret_cast<const v16h*>(bp + (size_t)(t0 + 1) * 512);
            acc = __builtin_amdgcn_wmma_f32_16x16x32_f16(false, a0, false, b0,
                                                         (short)0, acc, false, false);
            acc = __builtin_amdgcn_wmma_f32_16x16x32_f16(false, a1, false, b1,
                                                         (short)0, acc, false, false);
        }
    }

    const int n  = nblk * 16 + (lane & 15);      // out channel for this lane's column
    const float bv = bias[n];
    const int mofs = (lane >= 16) ? 8 : 0;
    float s = 0.f, s2 = 0.f;
    #pragma unroll
    for (int r = 0; r < 8; ++r) {
        float v = acc[r] + bv;
        const int idx2 = base + r + mofs;        // row M = r + mofs
        const int y2 = idx2 / W, x2 = idx2 % W;
        size_t off;
        if (SHUF) { // pixel_shuffle(2): n = c2*4 + oi*2 + oj
            const int c2 = n >> 2, oi = (n >> 1) & 1, oj = n & 1;
            off = (((size_t)b * (COUT / 4) + c2) * (2 * H) + (2 * y2 + oi)) * (size_t)(2 * W)
                  + (2 * x2 + oj);
        } else {
            off = (((size_t)b * COUT + n) * H + y2) * (size_t)W + x2;
        }
        raw[off] = v;
        s += v; s2 += v * v;
    }
    // GroupNorm(groups=2) stats; group index is wave-uniform for 16-aligned n blocks
    const int g = SHUF ? (n >> 7) : (n >> 5);
    for (int o = 16; o > 0; o >>= 1) { s += __shfl_down(s, o, 32); s2 += __shfl_down(s2, o, 32); }
    if (lane == 0) {
        const int bank = blockIdx.x & 63;        // 64-way banked to cut atomic contention
        atomicAdd(&stats[(((b * 2 + g) << 6) + bank) * 2 + 0], s);
        atomicAdd(&stats[(((b * 2 + g) << 6) + bank) * 2 + 1], s2);
    }
}

// fold 64 banks -> mean / rsqrt(var+eps) per (b, group)
__global__ void gn_finalize(const float* __restrict__ stats, float* __restrict__ mi, float cnt) {
    int i = threadIdx.x;
    if (i >= 32) return;
    float s = 0.f, s2 = 0.f;
    for (int k = 0; k < 64; ++k) {
        s  += stats[((i << 6) + k) * 2 + 0];
        s2 += stats[((i << 6) + k) * 2 + 1];
    }
    float m = s / cnt;
    float v = s2 / cnt - m * m;
    mi[2 * i]     = m;
    mi[2 * i + 1] = rsqrtf(v + 1e-5f);
}

// ---------------------------------------------------------------------------
// GN affine + SiLU (+ optional residual add), NCHW fp32 -> halo NHWC f16, via
// LDS-tiled transpose: coalesced reads AND writes. Block = 64 pixels x 64 ch.
// W is compile-time so halo store div/mod by W is a constant multiply, not a runtime divide.
// ---------------------------------------------------------------------------
template<bool ADD, int W>
__global__ __launch_bounds__(256)
void gn_apply_silu_t(const float* __restrict__ raw, const float* __restrict__ mi,
                     const float* __restrict__ gw, const float* __restrict__ gb,
                     const float* __restrict__ addsrc, half_t* __restrict__ out) {
    constexpr int HW = W * W;
    constexpr int WW = W + 2, HH = W + 2;
    __shared__ float tile[64][65];
    const int t = threadIdx.x;
    const int blocksPerImage = HW >> 6;
    const int b    = blockIdx.x / blocksPerImage;
    const int pix0 = (blockIdx.x % blocksPerImage) << 6;
    const int cr = t >> 6;     // 0..3
    const int pr = t & 63;
    const float m0 = mi[(b * 2 + 0) * 2], i0 = mi[(b * 2 + 0) * 2 + 1];
    const float m1 = mi[(b * 2 + 1) * 2], i1 = mi[(b * 2 + 1) * 2 + 1];
    #pragma unroll 4
    for (int it = 0; it < 16; ++it) {
        const int c = it * 4 + cr;
        const size_t idx = ((size_t)b * 64 + c) * HW + pix0 + pr;
        float v = raw[idx];
        v = (v - (c < 32 ? m0 : m1)) * (c < 32 ? i0 : i1) * gw[c] + gb[c];
        float a = v / (1.f + __expf(-v));
        if (ADD) a += addsrc[idx];
        tile[c][pr] = a;
    }
    __syncthreads();
    #pragma unroll 4
    for (int it = 0; it < 16; ++it) {
        const int flat = it * 256 + t;
        const int c  = flat & 63;
        const int px = flat >> 6;
        const int pix = pix0 + px;
        const int y = pix / W, x = pix % W;
        out[(((size_t)b * HH + (y + 1)) * WW + (x + 1)) * 64 + c] = (half_t)tile[c][px];
    }
}

// dec3: GN + SiLU fused with 1x1 readout conv (64 -> 3). Writes Y (fp32 NCHW).
__global__ void gn_apply_readout(const float* __restrict__ raw, const float* __restrict__ mi,
                                 const float* __restrict__ gw, const float* __restrict__ gb,
                                 const float* __restrict__ rw, const float* __restrict__ rb,
                                 float* __restrict__ Y) {
    const int H = 160, W = 160, C = 64;
    size_t tid = (size_t)blockIdx.x * blockDim.x + threadIdx.x;
    if (tid >= (size_t)16 * H * W) return;
    int x = (int)(tid % W);
    int y = (int)((tid / W) % H);
    int b = (int)(tid / ((size_t)H * W));
    float y0 = rb[0], y1 = rb[1], y2 = rb[2];
    float m0 = mi[(b * 2 + 0) * 2], i0 = mi[(b * 2 + 0) * 2 + 1];
    float m1 = mi[(b * 2 + 1) * 2], i1 = mi[(b * 2 + 1) * 2 + 1];
    for (int c = 0; c < C; ++c) {
        float mean = (c < 32) ? m0 : m1;
        float inv  = (c < 32) ? i0 : i1;
        float v = (raw[(((size_t)b * C + c) * H + y) * W + x] - mean) * inv * gw[c] + gb[c];
        float a = v / (1.f + __expf(-v));
        y0 += a * rw[c]; y1 += a * rw[64 + c]; y2 += a * rw[128 + c];
    }
    size_t o = (size_t)b * 3 * H * W + (size_t)y * W + x;
    Y[o] = y0; Y[o + H * W] = y1; Y[o + 2 * H * W] = y2;
}

// feamap conv (stride 4)/16 fused with unfold_p: patches[b][c][p][u]
__global__ void patches_kernel(const float* __restrict__ Y, const float* __restrict__ fw,
                               float* __restrict__ patches) {
    int tid = blockIdx.x * blockDim.x + threadIdx.x;
    if (tid >= 16 * 3 * 100 * 16) return;
    int u = tid & 15;
    int p = (tid >> 4) % 100;
    int c = (tid / 1600) % 3;
    int b = tid / 4800;
    int hi = u >> 2, wi = u & 3;
    int pi = p / 10, pj = p % 10;
    int y0 = 4 * (hi * 10 + pi), x0 = 4 * (wi * 10 + pj);
    float s = 0.f;
    for (int ci = 0; ci < 3; ++ci)
        for (int i = 0; i < 4; ++i)
            for (int j = 0; j < 4; ++j)
                s += Y[(((size_t)b * 3 + ci) * 160 + (y0 + i)) * 160 + (x0 + j)]
                     * fw[((c * 3 + ci) * 4 + i) * 4 + j];
    patches[tid] = s * (1.f / 16.f);
}

__global__ void count_nz(const float* __restrict__ attn, float* __restrict__ invnz) {
    int tid = blockIdx.x * blockDim.x + threadIdx.x;
    if (tid >= 16 * 3 * 256) return;
    const float* p = attn + (size_t)tid * 16;
    int cnt = 0;
    for (int k = 0; k < 16; ++k) cnt += (p[k] != 0.f);
    invnz[tid] = 1.f / ((float)cnt + 1e-5f);
}

// att[b,c,l,p] = invnz[b,c,l] * sum_k attn[b,c,l,k] * patches[b,c,p,k]
__global__ void att_kernel(const float* __restrict__ attn, const float* __restrict__ invnz,
                           const float* __restrict__ patches, float* __restrict__ att) {
    size_t tid = (size_t)blockIdx.x * blockDim.x + threadIdx.x;
    if (tid >= (size_t)16 * 3 * 256 * 100) return;
    int p = (int)(tid % 100);
    int l = (int)((tid / 100) % 256);
    int bc = (int)(tid / 25600);
    const float* ar = attn + ((size_t)bc * 256 + l) * 16;
    const float* pr = patches + ((size_t)bc * 100 + p) * 16;
    float s = 0.f;
    #pragma unroll
    for (int k = 0; k < 16; ++k) s += ar[k] * pr[k];
    att[tid] = s * invnz[bc * 256 + l];
}

// fold_p + final: out = Y * (1 + correction), in place on d_out
__global__ void final_kernel(float* __restrict__ Y, const float* __restrict__ att) {
    size_t tid = (size_t)blockIdx.x * blockDim.x + threadIdx.x;
    if (tid >= (size_t)16 * 3 * 160 * 160) return;
    int x = (int)(tid % 160);
    int y = (int)((tid / 160) % 160);
    int bc = (int)(tid / 25600);
    int l = (y / 10) * 16 + (x / 10);
    int p = (y % 10) * 10 + (x % 10);
    float corr = att[((size_t)bc * 256 + l) * 100 + p];
    Y[tid] *= (1.f + corr);
}

static inline unsigned cdiv(size_t a, unsigned b) { return (unsigned)((a + b - 1) / b); }

extern "C" void kernel_launch(void* const* d_in, const int* in_sizes, int n_in,
                              void* d_out, int out_size, void* d_ws, size_t ws_size,
                              hipStream_t stream) {
    const float* attn    = (const float*)d_in[0];
    const float* hid     = (const float*)d_in[1];
    const float* enc1    = (const float*)d_in[2];
    const float* dec0_w  = (const float*)d_in[3];  const float* dec0_b  = (const float*)d_in[4];
    const float* dec0_gw = (const float*)d_in[5];  const float* dec0_gb = (const float*)d_in[6];
    const float* dec1_w  = (const float*)d_in[7];  const float* dec1_b  = (const float*)d_in[8];
    const float* dec1_gw = (const float*)d_in[9];  const float* dec1_gb = (const float*)d_in[10];
    const float* dec2_w  = (const float*)d_in[11]; const float* dec2_b  = (const float*)d_in[12];
    const float* dec2_gw = (const float*)d_in[13]; const float* dec2_gb = (const float*)d_in[14];
    const float* dec3_w  = (const float*)d_in[15]; const float* dec3_b  = (const float*)d_in[16];
    const float* dec3_gw = (const float*)d_in[17]; const float* dec3_gb = (const float*)d_in[18];
    const float* ro_w    = (const float*)d_in[19]; const float* ro_b    = (const float*)d_in[20];
    const float* fm_w    = (const float*)d_in[21];
    float* Y = (float*)d_out;

    char* ws = (char*)d_ws;
    size_t o = 0;
    auto alloc = [&](size_t bytes) { size_t r = o; o = (o + bytes + 255) & ~(size_t)255; return r; };
    size_t bp0 = alloc((size_t)256 * 576 * 2);
    size_t bp1 = alloc((size_t)64  * 576 * 2);
    size_t bp2 = alloc((size_t)256 * 576 * 2);
    size_t bp3 = alloc((size_t)64  * 576 * 2);
    // halo-padded NHWC f16 staging (contiguous region, zeroed each call)
    size_t in0 = alloc((size_t)16 * 42 * 42 * 64 * 2);      // hid
    size_t a0  = alloc((size_t)16 * 82 * 82 * 64 * 2);      // act0
    size_t a1  = alloc((size_t)16 * 82 * 82 * 64 * 2);      // act1
    size_t a2  = alloc((size_t)16 * 162 * 162 * 64 * 2);    // act2 + enc1
    size_t stageEnd = o;
    size_t rA  = alloc((size_t)16 * 64 * 160 * 160 * 4);    // raw conv0 / conv2
    size_t rB  = alloc((size_t)16 * 64 * 160 * 160 * 4);    // raw conv1 / conv3
    size_t st  = alloc((size_t)4 * 4096 * sizeof(float));   // [layer][b*2][64 banks][2]
    size_t sf  = alloc((size_t)4 * 64 * sizeof(float));     // [layer][b*2][2] mean,inv
    size_t pt  = alloc((size_t)16 * 3 * 100 * 16 * 4);
    size_t nz  = alloc((size_t)16 * 3 * 256 * 4);
    size_t at  = alloc((size_t)16 * 3 * 256 * 100 * 4);

    // zero halo staging + GN stat accumulators (single pass, every call)
    hipMemsetAsync(ws + in0, 0, stageEnd - in0, stream);
    hipMemsetAsync(ws + st, 0, (size_t)4 * 4096 * sizeof(float), stream);

    // weight prep
    pack_weights<<<cdiv(256 * 576, 256), 256, 0, stream>>>(dec0_w, (half_t*)(ws + bp0), 256);
    pack_weights<<<cdiv(64  * 576, 256), 256, 0, stream>>>(dec1_w, (half_t*)(ws + bp1), 64);
    pack_weights<<<cdiv(256 * 576, 256), 256, 0, stream>>>(dec2_w, (half_t*)(ws + bp2), 256);
    pack_weights<<<cdiv(64  * 576, 256), 256, 0, stream>>>(dec3_w, (half_t*)(ws + bp3), 64);
    nchw_to_nhwc_f16_halo<<<cdiv((size_t)16 * 64 * 1600, 256), 256, 0, stream>>>(
        hid, (half_t*)(ws + in0), 16, 64, 40, 40);

    float* stL = (float*)(ws + st);
    float* miL = (float*)(ws + sf);

    // dec0: conv(64->256) + shuffle -> [16,64,80,80]
    conv3x3_wmma<256, 40, 40, true><<<dim3(16 * 1600 / 16, 4), 128, 0, stream>>>(
        (const half_t*)(ws + in0), (const half_t*)(ws + bp0), dec0_b, (float*)(ws + rA), stL);
    gn_finalize<<<1, 32, 0, stream>>>(stL, miL, 32.f * 80 * 80);
    gn_apply_silu_t<false, 80><<<16 * 6400 / 64, 256, 0, stream>>>(
        (const float*)(ws + rA), miL, dec0_gw, dec0_gb, nullptr, (half_t*)(ws + a0));

    // dec1: conv(64->64) -> [16,64,80,80]
    conv3x3_wmma<64, 80, 80, false><<<dim3(16 * 6400 / 16, 1), 128, 0, stream>>>(
        (const half_t*)(ws + a0), (const half_t*)(ws + bp1), dec1_b, (float*)(ws + rB),
        stL + 4096);
    gn_finalize<<<1, 32, 0, stream>>>(stL + 4096, miL + 64, 32.f * 80 * 80);
    gn_apply_silu_t<false, 80><<<16 * 6400 / 64, 256, 0, stream>>>(
        (const float*)(ws + rB), miL + 64, dec1_gw, dec1_gb, nullptr, (half_t*)(ws + a1));

    // dec2: conv(64->256) + shuffle -> [16,64,160,160]; apply adds enc1 for dec3 input
    conv3x3_wmma<256, 80, 80, true><<<dim3(16 * 6400 / 16, 4), 128, 0, stream>>>(
        (const half_t*)(ws + a1), (const half_t*)(ws + bp2), dec2_b, (float*)(ws + rA),
        stL + 2 * 4096);
    gn_finalize<<<1, 32, 0, stream>>>(stL + 2 * 4096, miL + 128, 32.f * 160 * 160);
    gn_apply_silu_t<true, 160><<<16 * 25600 / 64, 256, 0, stream>>>(
        (const float*)(ws + rA), miL + 128, dec2_gw, dec2_gb, enc1, (half_t*)(ws + a2));

    // dec3: conv(64->64) -> raw; GN+SiLU fused with readout -> Y (d_out)
    conv3x3_wmma<64, 160, 160, false><<<dim3(16 * 25600 / 16, 1), 128, 0, stream>>>(
        (const half_t*)(ws + a2), (const half_t*)(ws + bp3), dec3_b, (float*)(ws + rB),
        stL + 3 * 4096);
    gn_finalize<<<1, 32, 0, stream>>>(stL + 3 * 4096, miL + 192, 32.f * 160 * 160);
    gn_apply_readout<<<cdiv((size_t)16 * 160 * 160, 256), 256, 0, stream>>>(
        (const float*)(ws + rB), miL + 192, dec3_gw, dec3_gb, ro_w, ro_b, Y);

    // attention correction tail
    patches_kernel<<<cdiv(16 * 3 * 100 * 16, 256), 256, 0, stream>>>(Y, fm_w, (float*)(ws + pt));
    count_nz<<<cdiv(16 * 3 * 256, 256), 256, 0, stream>>>(attn, (float*)(ws + nz));
    att_kernel<<<cdiv((size_t)16 * 3 * 256 * 100, 256), 256, 0, stream>>>(
        attn, (const float*)(ws + nz), (const float*)(ws + pt), (float*)(ws + at));
    final_kernel<<<cdiv((size_t)16 * 3 * 160 * 160, 256), 256, 0, stream>>>(
        Y, (const float*)(ws + at));
}
